// BiDAFAttention_60395830116834
// MI455X (gfx1250) — matrix operand order repaired and checked by reading, not compiled, not run on hardware
//
#include <hip/hip_runtime.h>
#include <hip/hip_bf16.h>
#include <hip/hip_fp16.h>

// ---------------------------------------------------------------------------
// BiDAF attention for MI455X (gfx1250, wave32, WMMA f16->f32, TDM staging)
// B=64, CL=1024, QL=128, D=512, out = (B, CL, 4D) f32
// ---------------------------------------------------------------------------

typedef _Float16 v16h __attribute__((ext_vector_type(16)));
typedef _Float16 v8h  __attribute__((ext_vector_type(8)));
typedef float    v8f  __attribute__((ext_vector_type(8)));
typedef unsigned int v4u __attribute__((ext_vector_type(4)));
typedef unsigned int v8u __attribute__((ext_vector_type(8)));

#define B_    64
#define CL_   1024
#define QL_   128
#define D_    512
#define OD_   2048
#define NEGI  (-1e30f)

// ---- WMMA helper ----------------------------------------------------------

static __device__ __forceinline__ v8f wmma16(v16h a, v16h b, v8f c) {
  // D = A(16x32 f16) x B(32x16 f16) + C(16x16 f32)
  return __builtin_amdgcn_wmma_f32_16x16x32_f16(
      /*neg_a=*/false, a, /*neg_b=*/false, b,
      /*c_mod=*/(short)0, c, /*reuse_a=*/false, /*reuse_b=*/false);
}

// ---- Tensor Data Mover: 2D tile Global -> LDS (CDNA5 ISA, D# groups) ----
// group0: count=1 | lds_addr | global_addr[56:0] | type=2
// group1: data_size | tensor_dim0/1 | tile_dim0/1 | tensor_dim0_stride
static __device__ __forceinline__ void tdm_load_2d(
    void* lds_ptr, const void* gptr, unsigned dsz_log2,
    unsigned tensor_d0, unsigned tile_d0, unsigned tile_d1, unsigned stride0) {
  const unsigned long long ga = (unsigned long long)(size_t)gptr;
  const unsigned lds = (unsigned)(size_t)lds_ptr;  // flat low 32 = LDS offset
  v4u g0;
  g0[0] = 1u;                                         // count=1 (valid user D#)
  g0[1] = lds;                                        // lds_addr
  g0[2] = (unsigned)ga;                               // global_addr[31:0]
  g0[3] = (unsigned)((ga >> 32) & 0x01ffffffu) | (2u << 30);  // [56:32] | type=2
  v8u g1;
  g1[0] = dsz_log2 << 16;                             // data_size (log2 bytes)
  g1[1] = (tensor_d0 & 0xffffu) << 16;                // tensor_dim0[15:0]
  g1[2] = ((tensor_d0 >> 16) & 0xffffu) |             // tensor_dim0[31:16]
          ((tile_d1 & 0xffffu) << 16);                // tensor_dim1[15:0]
  g1[3] = ((tile_d1 >> 16) & 0xffffu) |               // tensor_dim1[31:16]
          ((tile_d0 & 0xffffu) << 16);                // tile_dim0
  g1[4] = tile_d1 & 0xffffu;                          // tile_dim1 (tile_dim2=0)
  g1[5] = stride0;                                    // tensor_dim0_stride[31:0]
  g1[6] = 0u;
  g1[7] = 0u;
  asm volatile("tensor_load_to_lds %0, %1" :: "s"(g0), "s"(g1) : "memory");
}

// ---- Fragment loaders (contiguous k = 16h + s, same for A and B) --------------------------------
// A 16x32 f16: lane<16 K=0..15, lane>=16 K=16..31, M=lane&15.
static __device__ __forceinline__ v16h load_a_rm(const _Float16* p, int ld) {
  const int lane = threadIdx.x & 31;
  const int r  = lane & 15;
  const int k0 = (lane & 16) ? 16 : 0;
  const _Float16* q0 = p + (size_t)r * ld + k0;
  v8h lo = *(const v8h*)(q0);
  v8h hi = *(const v8h*)(q0 + 8);
  v16h f;
#pragma unroll
  for (int t = 0; t < 8; ++t) { f[t] = lo[t]; f[t + 8] = hi[t]; }
  return f;
}

// A fragment built from f32 data scaled per-K (cq_weight), converted to f16.
static __device__ __forceinline__ v16h load_a_f32_scaled(const float* p, int ld,
                                                         const float* sc) {
  const int lane = threadIdx.x & 31;
  const int r  = lane & 15;
  const int k0 = (lane & 16) ? 16 : 0;
  const float* q0 = p + (size_t)r * ld + k0;
  v16h f;
#pragma unroll
  for (int t = 0; t < 8; ++t) {
    f[t]     = (_Float16)(q0[t]      * sc[k0 + t]);
    f[t + 8] = (_Float16)(q0[t + 8] * sc[k0 + 8 + t]);
  }
  return f;
}

// B 32x16 f16 from N-major storage (row n contiguous in K == B col-major).
// lanes 0-15: K=0..15; lanes 16-31: K=16..31; N = lane&15.
static __device__ __forceinline__ v16h load_b_nk(const _Float16* p, int ld) {
  const int lane = threadIdx.x & 31;
  const int n  = lane & 15;
  const int kb = (lane & 16) ? 16 : 0;
  const _Float16* q0 = p + (size_t)n * ld + kb;
  v8h lo = *(const v8h*)(q0);
  v8h hi = *(const v8h*)(q0 + 8);
  v16h f;
#pragma unroll
  for (int t = 0; t < 8; ++t) { f[t] = lo[t]; f[t + 8] = hi[t]; }
  return f;
}

// B 32x16 from K-major (row-major KxN) f16 storage: strided gather.
static __device__ __forceinline__ v16h load_b_kn(const _Float16* p, int ld) {
  const int lane = threadIdx.x & 31;
  const int n  = lane & 15;
  const int kb = (lane & 16) ? 16 : 0;
  v16h f;
#pragma unroll
  for (int t = 0; t < 16; ++t) f[t] = p[(size_t)(kb + t) * ld + n];
  return f;
}

// Same but f32 source (LDS tile staged by TDM), convert while gathering.
static __device__ __forceinline__ v16h load_b_kn_f32(const float* p, int ld) {
  const int lane = threadIdx.x & 31;
  const int n  = lane & 15;
  const int kb = (lane & 16) ? 16 : 0;
  v16h f;
#pragma unroll
  for (int t = 0; t < 16; ++t) f[t] = (_Float16)p[(size_t)(kb + t) * ld + n];
  return f;
}

// ---- K0: per-row dot with weight vector (+ optional f16 cast of the rows) --

__global__ void k_rowdot(const float* __restrict__ x, const float* __restrict__ w,
                         float* __restrict__ dots, _Float16* __restrict__ xh) {
  const int row = blockIdx.x;
  const float* src = x + (size_t)row * D_;
  float acc = 0.f;
  for (int d = threadIdx.x; d < D_; d += 256) {
    float v = src[d];
    if (xh) xh[(size_t)row * D_ + d] = (_Float16)v;
    acc += v * w[d];
  }
  __shared__ float red[8];
#pragma unroll
  for (int off = 16; off; off >>= 1) acc += __shfl_xor(acc, off, 32);
  if ((threadIdx.x & 31) == 0) red[threadIdx.x >> 5] = acc;
  __syncthreads();
  if (threadIdx.x == 0) {
    float t = red[0];
#pragma unroll
    for (int k = 1; k < 8; ++k) t += red[k];
    dots[row] = t;
  }
}

// ---- K1: s[b,i,j] = (c*w_cq).q + cw[i] + qw[j] + bias  (WMMA) --------------

__global__ void __launch_bounds__(256)
k_sgemm(const float* __restrict__ c, const _Float16* __restrict__ qh,
        const float* __restrict__ cqw_g, const float* __restrict__ cw,
        const float* __restrict__ qw, const float* __restrict__ bias,
        float* __restrict__ s_raw) {
  const int b     = blockIdx.x >> 3;
  const int ibase = (blockIdx.x & 7) * 128;

  __shared__ float cqw[D_];
  for (int d = threadIdx.x; d < D_; d += 256) cqw[d] = cqw_g[d];
  __syncthreads();

  const float*    c_b  = c  + ((size_t)b * CL_ + ibase) * D_;
  const _Float16* qh_b = qh + (size_t)b * QL_ * D_;

  const int w     = threadIdx.x >> 5;
  const int mbase = w * 16;

  v8f acc[8] = {};
  for (int kk = 0; kk < D_; kk += 32) {
    // gather all fragments first -> many loads in flight before the waits
    v16h bf[8];
#pragma unroll
    for (int nt = 0; nt < 8; ++nt)
      bf[nt] = load_b_nk(qh_b + (size_t)(nt * 16) * D_ + kk, D_);
    v16h af = load_a_f32_scaled(c_b + (size_t)mbase * D_ + kk, D_, cqw + kk);
    if (kk + 32 < D_)
      __builtin_prefetch(c_b + (size_t)mbase * D_ + kk + 32, 0, 0);
#pragma unroll
    for (int nt = 0; nt < 8; ++nt) acc[nt] = wmma16(af, bf[nt], acc[nt]);
  }

  const int lane = threadIdx.x & 31;
  const int n  = lane & 15;
  const int mh = (lane & 16) ? 8 : 0;
  const float bs = bias[0];
  float* s_b = s_raw + ((size_t)b * CL_ + ibase) * QL_;
#pragma unroll
  for (int nt = 0; nt < 8; ++nt) {
    const int j = nt * 16 + n;
    const float qwv = qw[b * QL_ + j] + bs;
#pragma unroll
    for (int r = 0; r < 8; ++r) {
      const int m = mbase + mh + r;
      s_b[(size_t)m * QL_ + j] = acc[nt][r] + cw[b * CL_ + ibase + m] + qwv;
    }
  }
}

// ---- K2: masked softmax over j (QL) -> s1 f16; one wave32 per row ----------

__global__ void k_softmax_rows(const float* __restrict__ s_raw,
                               const int* __restrict__ q_mask,
                               _Float16* __restrict__ s1h) {
  const int row  = blockIdx.x * 8 + (threadIdx.x >> 5);  // b*CL + i
  const int b    = row >> 10;
  const int lane = threadIdx.x & 31;
  const float* src = s_raw + (size_t)row * QL_;
  const int* qm = q_mask + b * QL_;

  float v[4];
  float mx = NEGI;
#pragma unroll
  for (int t = 0; t < 4; ++t) {
    const int j = lane + t * 32;
    v[t] = qm[j] ? src[j] : NEGI;
    mx = fmaxf(mx, v[t]);
  }
#pragma unroll
  for (int off = 16; off; off >>= 1) mx = fmaxf(mx, __shfl_xor(mx, off, 32));
  float sum = 0.f;
#pragma unroll
  for (int t = 0; t < 4; ++t) { v[t] = __expf(v[t] - mx); sum += v[t]; }
#pragma unroll
  for (int off = 16; off; off >>= 1) sum += __shfl_xor(sum, off, 32);
  const float inv = 1.0f / sum;
  _Float16* dst = s1h + (size_t)row * QL_;
#pragma unroll
  for (int t = 0; t < 4; ++t) dst[lane + t * 32] = (_Float16)(v[t] * inv);
}

// ---- K3: masked softmax over i (CL) -> s2^T f16 (transposed, coalesced) ----

__global__ void k_softmax_cols(const float* __restrict__ s_raw,
                               const int* __restrict__ c_mask,
                               _Float16* __restrict__ s2th) {
  const int b = blockIdx.x >> 7;
  const int j = blockIdx.x & 127;
  const float* s_b = s_raw + (size_t)b * CL_ * QL_ + j;
  const int* cm = c_mask + b * CL_;

  __shared__ float red[8];
  __shared__ float bval;

  float v[4];
  float mx = NEGI;
#pragma unroll
  for (int t = 0; t < 4; ++t) {
    const int i = threadIdx.x + t * 256;
    v[t] = cm[i] ? s_b[(size_t)i * QL_] : NEGI;
    mx = fmaxf(mx, v[t]);
  }
#pragma unroll
  for (int off = 16; off; off >>= 1) mx = fmaxf(mx, __shfl_xor(mx, off, 32));
  if ((threadIdx.x & 31) == 0) red[threadIdx.x >> 5] = mx;
  __syncthreads();
  if (threadIdx.x == 0) {
    float m = red[0];
#pragma unroll
    for (int k = 1; k < 8; ++k) m = fmaxf(m, red[k]);
    bval = m;
  }
  __syncthreads();
  mx = bval;
  float sum = 0.f;
#pragma unroll
  for (int t = 0; t < 4; ++t) { v[t] = __expf(v[t] - mx); sum += v[t]; }
#pragma unroll
  for (int off = 16; off; off >>= 1) sum += __shfl_xor(sum, off, 32);
  __syncthreads();
  if ((threadIdx.x & 31) == 0) red[threadIdx.x >> 5] = sum;
  __syncthreads();
  if (threadIdx.x == 0) {
    float s = 0.f;
#pragma unroll
    for (int k = 0; k < 8; ++k) s += red[k];
    bval = 1.0f / s;
  }
  __syncthreads();
  const float inv = bval;
  _Float16* dst = s2th + ((size_t)b * QL_ + j) * (size_t)CL_;
#pragma unroll
  for (int t = 0; t < 4; ++t)
    dst[threadIdx.x + t * 256] = (_Float16)(v[t] * inv);
}

// ---- K4: t[b,j,d] = sum_i s2^T[b,j,i]*c[b,i,d]  (WMMA, TDM double-buffer) --

__global__ void __launch_bounds__(256)
k_tgemm(const float* __restrict__ c, const _Float16* __restrict__ s2th,
        _Float16* __restrict__ th) {
  __shared__ float ctile[2][32 * 128];   // two 16 KB f32 tiles (double buffer)

  const int b  = blockIdx.x >> 2;
  const int dt = (blockIdx.x & 3) * 128;

  const float*    c_b = c + (size_t)b * CL_ * D_;
  const _Float16* a_b = s2th + (size_t)b * QL_ * CL_;

  const int w     = threadIdx.x >> 5;
  const int mbase = w * 16;
  const bool issuer = threadIdx.x < 32;  // wave 0 drives the TDM

  if (issuer)
    tdm_load_2d(&ctile[0][0], c_b + dt, /*4B*/2u, D_, 128u, 32u, D_);

  v8f acc[8] = {};
  for (int kk = 0; kk < CL_; kk += 32) {
    const int cur = (kk >> 5) & 1;
    if (issuer) __builtin_amdgcn_s_wait_tensorcnt(0);
    __syncthreads();                      // tile[cur] ready for everyone
    if (issuer && (kk + 32 < CL_))        // overlap next DMA with compute
      tdm_load_2d(&ctile[cur ^ 1][0], c_b + (size_t)(kk + 32) * D_ + dt,
                  2u, D_, 128u, 32u, D_);

    v16h af = load_a_rm(a_b + (size_t)mbase * CL_ + kk, CL_);
    v16h bf[8];
#pragma unroll
    for (int nt = 0; nt < 8; ++nt)
      bf[nt] = load_b_kn_f32(&ctile[cur][0] + nt * 16, 128);
#pragma unroll
    for (int nt = 0; nt < 8; ++nt) acc[nt] = wmma16(af, bf[nt], acc[nt]);
  }

  const int lane = threadIdx.x & 31;
  const int n  = lane & 15;
  const int mh = (lane & 16) ? 8 : 0;
  _Float16* t_b = th + (size_t)b * QL_ * D_;
#pragma unroll
  for (int nt = 0; nt < 8; ++nt)
#pragma unroll
    for (int r = 0; r < 8; ++r)
      t_b[(size_t)(mbase + mh + r) * D_ + dt + nt * 16 + n] = (_Float16)acc[nt][r];
}

// ---- K5: fused a = s1*q, bb = s1*t, out = [c, a, c*a, c*bb] ----------------

__global__ void __launch_bounds__(256)
k_out(const float* __restrict__ c, const _Float16* __restrict__ qh,
      const _Float16* __restrict__ th, const _Float16* __restrict__ s1h,
      float* __restrict__ out) {
  extern __shared__ _Float16 smem[];          // 2 * 128*512 f16 = 256 KB
  _Float16* q_s = smem;
  _Float16* t_s = smem + QL_ * D_;

  const int b     = blockIdx.x >> 3;
  const int ibase = (blockIdx.x & 7) * 128;

  // stage q[b] and t[b] panels via the Tensor Data Mover
  if (threadIdx.x < 32) {
    tdm_load_2d(q_s, qh + (size_t)b * QL_ * D_, /*2B*/1u, D_, D_, QL_, D_);
    tdm_load_2d(t_s, th + (size_t)b * QL_ * D_, 1u, D_, D_, QL_, D_);
    __builtin_amdgcn_s_wait_tensorcnt(0);
  }
  __syncthreads();

  const int w     = threadIdx.x >> 5;
  const int mbase = w * 16;

  // A fragments: 16 rows of s1 (K = QL = 128, four 16x32 fragments)
  const _Float16* a_b = s1h + ((size_t)b * CL_ + ibase + mbase) * QL_;
  v16h af[4];
#pragma unroll
  for (int kt = 0; kt < 4; ++kt) af[kt] = load_a_rm(a_b + kt * 32, QL_);

  const int lane = threadIdx.x & 31;
  const int n  = lane & 15;
  const int mh = (lane & 16) ? 8 : 0;
  const float* c_b   = c   + ((size_t)b * CL_ + ibase + mbase) * D_;
  float*       out_b = out + ((size_t)b * CL_ + ibase + mbase) * OD_;

  for (int nt = 0; nt < 32; ++nt) {
    const int dbase = nt * 16;
    v16h bq[4], bt[4];
#pragma unroll
    for (int kt = 0; kt < 4; ++kt) {
      bq[kt] = load_b_kn(q_s + (size_t)(kt * 32) * D_ + dbase, D_);
      bt[kt] = load_b_kn(t_s + (size_t)(kt * 32) * D_ + dbase, D_);
    }
    v8f aa = {};
    v8f bb = {};
#pragma unroll
    for (int kt = 0; kt < 4; ++kt) {       // two independent WMMA chains
      aa = wmma16(af[kt], bq[kt], aa);
      bb = wmma16(af[kt], bt[kt], bb);
    }
#pragma unroll
    for (int r = 0; r < 8; ++r) {
      const int m = mh + r;
      const float cv = c_b[(size_t)m * D_ + dbase + n];
      const float av = aa[r];
      const float bv = bb[r];
      float* orow = out_b + (size_t)m * OD_ + dbase + n;
      orow[0]      = cv;
      orow[D_]     = av;
      orow[2 * D_] = cv * av;
      orow[3 * D_] = cv * bv;
    }
  }
}

// ---------------------------------------------------------------------------

extern "C" void kernel_launch(void* const* d_in, const int* in_sizes, int n_in,
                              void* d_out, int out_size, void* d_ws, size_t ws_size,
                              hipStream_t stream) {
  (void)in_sizes; (void)n_in; (void)out_size; (void)ws_size;
  const float* c        = (const float*)d_in[0];
  const float* q        = (const float*)d_in[1];
  const int*   c_mask   = (const int*)d_in[2];
  const int*   q_mask   = (const int*)d_in[3];
  const float* c_weight = (const float*)d_in[4];
  const float* q_weight = (const float*)d_in[5];
  const float* cq_weight= (const float*)d_in[6];
  const float* bias     = (const float*)d_in[7];
  float* out = (float*)d_out;

  char* ws = (char*)d_ws;
  size_t off = 0;
  float*    s_raw = (float*)(ws + off);    off += (size_t)B_ * CL_ * QL_ * 4;  // 33.5 MB
  _Float16* s1h   = (_Float16*)(ws + off); off += (size_t)B_ * CL_ * QL_ * 2;  // 16.8 MB
  _Float16* s2th  = (_Float16*)(ws + off); off += (size_t)B_ * QL_ * CL_ * 2;  // 16.8 MB
  _Float16* qh    = (_Float16*)(ws + off); off += (size_t)B_ * QL_ * D_  * 2;  //  8.4 MB
  _Float16* th    = (_Float16*)(ws + off); off += (size_t)B_ * QL_ * D_  * 2;  //  8.4 MB
  float*    cw    = (float*)(ws + off);    off += (size_t)B_ * CL_ * 4;
  float*    qw    = (float*)(ws + off);    off += (size_t)B_ * QL_ * 4;

  k_rowdot<<<B_ * QL_, 256, 0, stream>>>(q, q_weight, qw, qh);
  k_rowdot<<<B_ * CL_, 256, 0, stream>>>(c, c_weight, cw, nullptr);
  k_sgemm<<<B_ * 8, 256, 0, stream>>>(c, qh, cq_weight, cw, qw, bias, s_raw);
  k_softmax_rows<<<B_ * CL_ / 8, 256, 0, stream>>>(s_raw, q_mask, s1h);
  k_softmax_cols<<<B_ * QL_, 256, 0, stream>>>(s_raw, c_mask, s2th);
  k_tgemm<<<B_ * 4, 256, 0, stream>>>(c, s2th, th);
  const size_t smem = (size_t)2 * QL_ * D_ * sizeof(_Float16);  // 256 KB (<= 320 KB WGP LDS)
  k_out<<<B_ * 8, 256, smem, stream>>>(c, qh, th, s1h, out);
}
